// DepthSeparableConv2d_44830868635691
// MI455X (gfx1250) — 8-bit matrix operand order repaired and checked by reading, compiled, not run on hardware
//
#include <hip/hip_runtime.h>
#include <stdint.h>

typedef __attribute__((ext_vector_type(8))) int v8i;
typedef signed char i8;

#define B_    32
#define CIN   128
#define COUT  256
#define H_    56
#define W_    56
#define HW    (H_ * W_)
#define P_    (B_ * HW)
#define K1    (CIN * 9)
#define ACC1_N (P_ * CIN)
#define ACC2_N (P_ * COUT)
#define EPS   1e-5f
#define PXB   64

#define O_ACC1  ((size_t)0)
#define O_ACC2  ((size_t)ACC1_N * 4)
#define O_X8    (O_ACC2 + (size_t)ACC2_N * 4)
#define O_H8    (O_X8 + (size_t)ACC1_N)
#define O_W1P   (O_H8 + (size_t)ACC1_N)
#define O_W2P   (O_W1P + (size_t)(CIN * K1))
#define O_STATS (O_W2P + (size_t)(COUT * CIN))

__device__ __forceinline__ void wait_async0() {
#if __has_builtin(__builtin_amdgcn_s_wait_asynccnt)
  __builtin_amdgcn_s_wait_asynccnt(0);
#else
  asm volatile("s_wait_asynccnt 0x0" ::: "memory");
#endif
}

__device__ __forceinline__ void async_stage32(unsigned ldsoff, const i8* gsrc) {
  asm volatile("global_load_async_to_lds_b128 %0, %1, off\n\t"
               "global_load_async_to_lds_b128 %0, %1, off offset:16"
               :: "v"(ldsoff), "v"(gsrc) : "memory");
}

__device__ __forceinline__ v8i load_a_frag(const i8* wk) {
  int2 a01 = *reinterpret_cast<const int2*>(wk + 0);
  int2 a23 = *reinterpret_cast<const int2*>(wk + 8);
  int2 a45 = *reinterpret_cast<const int2*>(wk + 32);
  int2 a67 = *reinterpret_cast<const int2*>(wk + 40);
  v8i A; A[0] = a01.x; A[1] = a01.y; A[2] = a23.x; A[3] = a23.y;
         A[4] = a45.x; A[5] = a45.y; A[6] = a67.x; A[7] = a67.y;
  return A;
}

__device__ __forceinline__ v8i load_b_frag(const i8* bp) {
  int4 lo = *reinterpret_cast<const int4*>(bp);
  int4 hi = *reinterpret_cast<const int4*>(bp + 32);
  v8i B; B[0] = lo.x; B[1] = lo.y; B[2] = lo.z; B[3] = lo.w;
         B[4] = hi.x; B[5] = hi.y; B[6] = hi.z; B[7] = hi.w;
  return B;
}

#define WMMA_IU8(A, Bm, C) \
  __builtin_amdgcn_wmma_i32_16x16x64_iu8(true, (A), true, (Bm), (C), false, false)

__global__ void k_init(float* stats) {
  for (int i = threadIdx.x; i < 1552; i += 256) stats[i] = 0.0f;
}

__global__ void k_pack_x(const int* __restrict__ xv, i8* __restrict__ x8) {
  int idx = blockIdx.x * 256 + threadIdx.x;
  if (idx >= ACC1_N) return;
  int b  = idx / (CIN * HW);
  int c  = (idx / HW) & (CIN - 1);
  int hw = idx % HW;
  x8[(size_t)(b * HW + hw) * CIN + c] = (i8)xv[idx];
}

__global__ void k_pack_w1(const float* __restrict__ w, i8* __restrict__ w1p) {
  int o = blockIdx.x * 256 + threadIdx.x;
  if (o >= CIN * K1) return;
  int co = o / K1;
  int rem = o % K1;
  int rs = rem >> 7;
  int ci = rem & 127;
  w1p[o] = (i8)(int)w[(co * CIN + ci) * 9 + rs];
}

__global__ void k_pack_w2(const float* __restrict__ w, i8* __restrict__ w2p) {
  int o = blockIdx.x * 256 + threadIdx.x;
  if (o >= COUT * CIN) return;
  w2p[o] = (i8)(int)w[o];
}

__global__ void k_conv1(const i8* __restrict__ x8, const i8* __restrict__ w1p,
                        int* __restrict__ acc1, unsigned* __restrict__ maxp) {
  __shared__ alignas(16) i8 btile[PXB * 128];
  const int tid  = threadIdx.x;
  const int wave = tid >> 5;
  const int lane = tid & 31;
  const int px_base = blockIdx.x * PXB;
  const int co_base = wave * 16;

  const int p_local = tid >> 2;
  const int cb = (tid & 3) * 32;
  const int px = px_base + p_local;
  const int b  = px / HW;
  const int hw = px % HW;
  const int h  = hw / W_;
  const int w  = hw % W_;
  const unsigned ldsoff = (unsigned)(size_t)&btile[p_local * 128 + cb];

  const v8i zero8 = {0, 0, 0, 0, 0, 0, 0, 0};
  v8i acc0 = zero8, acc1v = zero8, acc2v = zero8, acc3v = zero8;
  const int colB = lane & 15;
  const int kB = (lane >= 16) ? 16 : 0;
  const int kA = (lane >= 16) ? 16 : 0;
  const i8* wbase = w1p + (size_t)(co_base + (lane & 15)) * K1;

  for (int rs = 0; rs < 9; ++rs) {
    const int r = rs / 3, s = rs % 3;
    const int hh = h + r - 1, ww = w + s - 1;
    __syncthreads();
    if (hh >= 0 && hh < H_ && ww >= 0 && ww < W_) {
      async_stage32(ldsoff, x8 + (size_t)(b * HW + hh * W_ + ww) * CIN + cb);
    } else {
      int4 z = {0, 0, 0, 0};
      *reinterpret_cast<int4*>(&btile[p_local * 128 + cb]) = z;
      *reinterpret_cast<int4*>(&btile[p_local * 128 + cb + 16]) = z;
    }
    wait_async0();
    __syncthreads();

    const i8* wrow = wbase + rs * 128;
    __builtin_prefetch(wrow + 128, 0, 1);
#pragma unroll
    for (int kk = 0; kk < 2; ++kk) {
      v8i A  = load_a_frag(wrow + kk * 64 + kA);
      v8i B0 = load_b_frag(&btile[(0 * 16 + colB) * 128 + kk * 64 + kB]);
      v8i B1 = load_b_frag(&btile[(1 * 16 + colB) * 128 + kk * 64 + kB]);
      v8i B2 = load_b_frag(&btile[(2 * 16 + colB) * 128 + kk * 64 + kB]);
      v8i B3 = load_b_frag(&btile[(3 * 16 + colB) * 128 + kk * 64 + kB]);
      acc0  = WMMA_IU8(A, B0, acc0);
      acc1v = WMMA_IU8(A, B1, acc1v);
      acc2v = WMMA_IU8(A, B2, acc2v);
      acc3v = WMMA_IU8(A, B3, acc3v);
    }
  }

  const int colD = lane & 15;
  const int rowD = co_base + ((lane >= 16) ? 8 : 0);
  unsigned m = 0;
  const v8i accs[4] = {acc0, acc1v, acc2v, acc3v};
#pragma unroll
  for (int t = 0; t < 4; ++t) {
    const int pxo = px_base + t * 16 + colD;
#pragma unroll
    for (int v2 = 0; v2 < 8; ++v2) {
      int val = accs[t][v2];
      acc1[(size_t)pxo * CIN + rowD + v2] = val;
      unsigned a = (unsigned)(val < 0 ? -val : val);
      m = m > a ? m : a;
    }
  }
  atomicMax(maxp, m);
}

__global__ void k_conv2(const i8* __restrict__ h8, const i8* __restrict__ w2p,
                        int* __restrict__ acc2, unsigned* __restrict__ maxp) {
  __shared__ alignas(16) i8 btile[PXB * 128];
  const int tid  = threadIdx.x;
  const int wave = tid >> 5;
  const int lane = tid & 31;
  const int px_base = blockIdx.x * PXB;
  const int co_base = blockIdx.y * 128 + wave * 16;

  const int p_local = tid >> 2;
  const int cb = (tid & 3) * 32;
  const unsigned ldsoff = (unsigned)(size_t)&btile[p_local * 128 + cb];
  async_stage32(ldsoff, h8 + (size_t)(px_base + p_local) * CIN + cb);
  wait_async0();
  __syncthreads();

  const v8i zero8 = {0, 0, 0, 0, 0, 0, 0, 0};
  v8i acc0 = zero8, acc1v = zero8, acc2v = zero8, acc3v = zero8;
  const int colB = lane & 15;
  const int kB = (lane >= 16) ? 16 : 0;
  const int kA = (lane >= 16) ? 16 : 0;
  const i8* wbase = w2p + (size_t)(co_base + (lane & 15)) * CIN;

#pragma unroll
  for (int kk = 0; kk < 2; ++kk) {
    v8i A  = load_a_frag(wbase + kk * 64 + kA);
    v8i B0 = load_b_frag(&btile[(0 * 16 + colB) * 128 + kk * 64 + kB]);
    v8i B1 = load_b_frag(&btile[(1 * 16 + colB) * 128 + kk * 64 + kB]);
    v8i B2 = load_b_frag(&btile[(2 * 16 + colB) * 128 + kk * 64 + kB]);
    v8i B3 = load_b_frag(&btile[(3 * 16 + colB) * 128 + kk * 64 + kB]);
    acc0  = WMMA_IU8(A, B0, acc0);
    acc1v = WMMA_IU8(A, B1, acc1v);
    acc2v = WMMA_IU8(A, B2, acc2v);
    acc3v = WMMA_IU8(A, B3, acc3v);
  }

  const int colD = lane & 15;
  const int rowD = co_base + ((lane >= 16) ? 8 : 0);
  unsigned m = 0;
  const v8i accs[4] = {acc0, acc1v, acc2v, acc3v};
#pragma unroll
  for (int t = 0; t < 4; ++t) {
    const int pxo = px_base + t * 16 + colD;
#pragma unroll
    for (int v2 = 0; v2 < 8; ++v2) {
      int val = accs[t][v2];
      acc2[(size_t)pxo * COUT + rowD + v2] = val;
      unsigned a = (unsigned)(val < 0 ? -val : val);
      m = m > a ? m : a;
    }
  }
  atomicMax(maxp, m);
}

__global__ void k_shift(float* S, int sBase, const float* eA, const float* eB) {
  unsigned umax = reinterpret_cast<unsigned*>(S)[sBase - 1];
  float rng = (float)umax;
  float bw = ceilf(log2f(rng));
  float shift = fmaxf(bw - 7.0f, 0.0f);
  S[sBase] = exp2f(-shift);
  S[sBase + 1] = *eA + *eB + shift;
}

__global__ void k_stats(const int* __restrict__ acc, const float* __restrict__ S,
                        int sBase, float* __restrict__ sum, float* __restrict__ ssq,
                        int C, int total, int perBlock) {
  __shared__ float ls[256];
  __shared__ float lq[256];
  int t = threadIdx.x;
  if (t < C) { ls[t] = 0.0f; lq[t] = 0.0f; }
  __syncthreads();
  float scale = S[sBase];
  float p2e = exp2f(S[sBase + 1]);
  int base = blockIdx.x * perBlock;
  for (int i = t; i < perBlock; i += 256) {
    int idx = base + i;
    if (idx < total) {
      float a = (float)acc[idx];
      float q = fminf(fmaxf(rintf(a * scale), -128.0f), 127.0f);
      float f = q * p2e;
      int c = idx & (C - 1);
      atomicAdd(&ls[c], f);
      atomicAdd(&lq[c], f * f);
    }
  }
  __syncthreads();
  if (t < C) { atomicAdd(&sum[t], ls[t]); atomicAdd(&ssq[t], lq[t]); }
}

__global__ void k_finalize(const float* sum, const float* ssq,
                           float* mean, float* istd, int C, float n) {
  int t = threadIdx.x;
  if (t >= C) return;
  float m = sum[t] / n;
  float v = ssq[t] / n - m * m;
  mean[t] = m;
  istd[t] = rsqrtf(v + EPS);
}

__global__ void k_bnmax(const int* __restrict__ acc, float* S, int sBase,
                        const float* __restrict__ mean, const float* __restrict__ istd,
                        const float* __restrict__ g, const float* __restrict__ bt,
                        int C, int total, int perBlock) {
  __shared__ unsigned lmax;
  if (threadIdx.x == 0) lmax = 0u;
  __syncthreads();
  float scale = S[sBase];
  float p2e = exp2f(S[sBase + 1]);
  unsigned m = 0;
  int base = blockIdx.x * perBlock;
  for (int i = threadIdx.x; i < perBlock; i += 256) {
    int idx = base + i;
    if (idx < total) {
      int c = idx & (C - 1);
      float a = (float)acc[idx];
      float q = fminf(fmaxf(rintf(a * scale), -128.0f), 127.0f);
      float f = q * p2e;
      float y = g[c] * (f - mean[c]) * istd[c] + bt[c];
      unsigned u = __float_as_uint(fabsf(y));
      m = m > u ? m : u;
    }
  }
  atomicMax(&lmax, m);
  __syncthreads();
  if (threadIdx.x == 0) atomicMax(reinterpret_cast<unsigned*>(S) + sBase + 2, lmax);
}

__global__ void k_qexp(float* S, int sBase) {
  float mx = __uint_as_float(reinterpret_cast<unsigned*>(S)[sBase + 2]);
  float bw = ceilf(log2f(mx));
  float e = bw - 7.0f;
  S[sBase + 3] = exp2f(-e);
  S[sBase + 4] = e;
}

__global__ void k_quant1(const int* __restrict__ acc, const float* __restrict__ S,
                         const float* __restrict__ mean, const float* __restrict__ istd,
                         const float* __restrict__ g, const float* __restrict__ bt,
                         i8* __restrict__ h8) {
  int idx = blockIdx.x * 256 + threadIdx.x;
  if (idx >= ACC1_N) return;
  int c = idx & (CIN - 1);
  float a = (float)acc[idx];
  float q = fminf(fmaxf(rintf(a * S[1]), -128.0f), 127.0f);
  float f = q * exp2f(S[2]);
  float y = g[c] * (f - mean[c]) * istd[c] + bt[c];
  float q2 = fminf(fmaxf(rintf(y * S[4]), -128.0f), 127.0f);
  q2 = fmaxf(q2, 0.0f);
  h8[idx] = (i8)q2;
}

__global__ void k_quant2(const int* __restrict__ acc, const float* __restrict__ S,
                         const float* __restrict__ mean, const float* __restrict__ istd,
                         const float* __restrict__ g, const float* __restrict__ bt,
                         float* __restrict__ out) {
  int idx = blockIdx.x * 256 + threadIdx.x;
  if (idx >= ACC2_N) return;
  int c  = idx & (COUT - 1);
  int px = idx >> 8;
  float a = (float)acc[idx];
  float q = fminf(fmaxf(rintf(a * S[7]), -128.0f), 127.0f);
  float f = q * exp2f(S[8]);
  float y = g[c] * (f - mean[c]) * istd[c] + bt[c];
  float q2 = fminf(fmaxf(rintf(y * S[10]), -128.0f), 127.0f);
  q2 = fmaxf(q2, 0.0f);
  int b  = px / HW;
  int hw = px % HW;
  out[((size_t)b * COUT + c) * HW + hw] = q2;
}

__global__ void k_write_e(float* out, const float* S) {
  out[ACC2_N] = S[11];
}

extern "C" void kernel_launch(void* const* d_in, const int* in_sizes, int n_in,
                              void* d_out, int out_size, void* d_ws, size_t ws_size,
                              hipStream_t stream) {
  const int*   x_val   = (const int*)d_in[0];
  const float* x_exp   = (const float*)d_in[1];
  const float* wDW     = (const float*)d_in[2];
  const float* wDW_exp = (const float*)d_in[3];
  const float* wPW     = (const float*)d_in[4];
  const float* wPW_exp = (const float*)d_in[5];
  const float* g1      = (const float*)d_in[6];
  const float* b1      = (const float*)d_in[7];
  const float* g2      = (const float*)d_in[8];
  const float* b2      = (const float*)d_in[9];

  char* ws = (char*)d_ws;
  int*  acc1 = (int*)(ws + O_ACC1);
  int*  acc2 = (int*)(ws + O_ACC2);
  i8*   x8   = (i8*)(ws + O_X8);
  i8*   h8   = (i8*)(ws + O_H8);
  i8*   w1p  = (i8*)(ws + O_W1P);
  i8*   w2p  = (i8*)(ws + O_W2P);
  float* S   = (float*)(ws + O_STATS);
  float* sum1  = S + 16;
  float* ssq1  = sum1 + 128;
  float* mean1 = ssq1 + 128;
  float* istd1 = mean1 + 128;
  float* sum2  = istd1 + 128;
  float* ssq2  = sum2 + 256;
  float* mean2 = ssq2 + 256;
  float* istd2 = mean2 + 256;
  float* out = (float*)d_out;

  k_init<<<1, 256, 0, stream>>>(S);
  k_pack_x<<<(ACC1_N + 255) / 256, 256, 0, stream>>>(x_val, x8);
  k_pack_w1<<<(CIN * K1 + 255) / 256, 256, 0, stream>>>(wDW, w1p);
  k_pack_w2<<<(COUT * CIN + 255) / 256, 256, 0, stream>>>(wPW, w2p);

  k_conv1<<<P_ / PXB, 256, 0, stream>>>(x8, w1p, acc1, (unsigned*)S + 0);
  k_shift<<<1, 1, 0, stream>>>(S, 1, x_exp, wDW_exp);
  k_stats<<<ACC1_N / 8192, 256, 0, stream>>>(acc1, S, 1, sum1, ssq1, CIN, ACC1_N, 8192);
  k_finalize<<<1, 256, 0, stream>>>(sum1, ssq1, mean1, istd1, CIN, (float)P_);
  k_bnmax<<<ACC1_N / 8192, 256, 0, stream>>>(acc1, S, 1, mean1, istd1, g1, b1, CIN, ACC1_N, 8192);
  k_qexp<<<1, 1, 0, stream>>>(S, 1);
  k_quant1<<<(ACC1_N + 255) / 256, 256, 0, stream>>>(acc1, S, mean1, istd1, g1, b1, h8);

  k_conv2<<<dim3(P_ / PXB, 2), 256, 0, stream>>>(h8, w2p, acc2, (unsigned*)S + 6);
  k_shift<<<1, 1, 0, stream>>>(S, 7, S + 5, wPW_exp);
  k_stats<<<ACC2_N / 8192, 256, 0, stream>>>(acc2, S, 7, sum2, ssq2, COUT, ACC2_N, 8192);
  k_finalize<<<1, 256, 0, stream>>>(sum2, ssq2, mean2, istd2, COUT, (float)P_);
  k_bnmax<<<ACC2_N / 8192, 256, 0, stream>>>(acc2, S, 7, mean2, istd2, g2, b2, COUT, ACC2_N, 8192);
  k_qexp<<<1, 1, 0, stream>>>(S, 7);
  k_quant2<<<(ACC2_N + 255) / 256, 256, 0, stream>>>(acc2, S, mean2, istd2, g2, b2, out);
  k_write_e<<<1, 1, 0, stream>>>(out, S);
}
